// DependencyNeuralModel_25529285607739
// MI455X (gfx1250) — matrix operand order repaired and checked by reading, not compiled, not run on hardware
//
#include <hip/hip_runtime.h>
#include <math.h>

typedef __attribute__((ext_vector_type(16))) _Float16 v16h;
typedef __attribute__((ext_vector_type(8)))  _Float16 v8h;
typedef __attribute__((ext_vector_type(8)))  float    v8f;

#define SLEN 512
#define HDIM 512
#define NBINS 17

__device__ __forceinline__ float sigf(float x) { return 1.0f / (1.0f + __expf(-x)); }

// ---------------------------------------------------------------------------
// Embedding gather -> f16, padded to 384 columns (300 word + 64 tag + 20 zero)
// ---------------------------------------------------------------------------
__global__ void embed_kernel(const int* __restrict__ words, const int* __restrict__ tags,
                             const float* __restrict__ we, const float* __restrict__ te,
                             _Float16* __restrict__ x0) {
    int s = blockIdx.x;
    int w = words[s], t = tags[s];
    for (int k = threadIdx.x; k < 384; k += blockDim.x) {
        float v = 0.0f;
        if (k < 300)      v = we[(size_t)w * 300 + k];
        else if (k < 364) v = te[(size_t)t * 64 + (k - 300)];
        x0[(size_t)s * 384 + k] = (_Float16)v;
    }
}

// f32 [rows,K] -> f16 [rows,Kp] zero-padded
__global__ void cvt_kernel(const float* __restrict__ src, _Float16* __restrict__ dst,
                           int K, int Kp) {
    size_t row = blockIdx.x;
    for (int k = threadIdx.x; k < Kp; k += blockDim.x)
        dst[row * Kp + k] = (k < K) ? (_Float16)src[row * K + k] : (_Float16)0.0f;
}

// combined LSTM bias: out = bih + bhh (covers both directions, n elements)
__global__ void bias_kernel(const float* __restrict__ a, const float* __restrict__ b,
                            float* __restrict__ o, int n) {
    int i = blockIdx.x * blockDim.x + threadIdx.x;
    if (i < n) o[i] = a[i] + b[i];
}

// ---------------------------------------------------------------------------
// Register-tiled WMMA GEMM:
//   C[M,N] (f32) = A[M,K] (f16,row-major) * B[N,K]^T (f16,row-major) + bias[N]
// Block = 256 threads = 8 waves (2 M x 4 N); each wave owns a 32x32 tile
// (2 A-frags x 2 B-frags -> 4 accumulators): 4 v_wmma per 8 b128 loads.
// Dims: M % 64 == 0, N % 128 == 0, K % 32 == 0.
// ---------------------------------------------------------------------------
__global__ __launch_bounds__(256)
void wmma_gemm(const _Float16* __restrict__ A, const _Float16* __restrict__ B,
               const float* __restrict__ bias, float* __restrict__ C,
               int K, int lda, int ldb, int ldc) {
    const int lane = threadIdx.x & 31;
    const int wv   = threadIdx.x >> 5;                  // 0..7
    const int m0   = blockIdx.y * 64  + (wv & 1) * 32;  // wave 32-row strip
    const int n0   = blockIdx.x * 128 + (wv >> 1) * 32; // wave 32-col strip
    const int hl   = lane >> 4;                         // lane half (0/1)
    const int l15  = lane & 15;
    const int akb  = hl * 16;                           // A K-subchunk base
    const int bkb  = hl * 16;                           // B K-subchunk base

    const _Float16* arow0 = A + (size_t)(m0 + l15) * lda;
    const _Float16* arow1 = arow0 + (size_t)16 * lda;
    const _Float16* brow0 = B + (size_t)(n0 + l15) * ldb;
    const _Float16* brow1 = brow0 + (size_t)16 * ldb;

    v8f acc00 = {}, acc01 = {}, acc10 = {}, acc11 = {};
    for (int k0 = 0; k0 < K; k0 += 32) {
        // A fragments: two contiguous 8-half loads per 16x32 tile
        v8h a0lo = *(const v8h*)(arow0 + k0 + akb);
        v8h a0hi = *(const v8h*)(arow0 + k0 + 8 + akb);
        v8h a1lo = *(const v8h*)(arow1 + k0 + akb);
        v8h a1hi = *(const v8h*)(arow1 + k0 + 8 + akb);
        v16h a0, a1;
#pragma unroll
        for (int i = 0; i < 8; ++i) {
            a0[i] = a0lo[i]; a0[8 + i] = a0hi[i];
            a1[i] = a1lo[i]; a1[8 + i] = a1hi[i];
        }
        // B fragments: one contiguous 16-half load per 32x16 tile
        v16h b0 = *(const v16h*)(brow0 + k0 + bkb);
        v16h b1 = *(const v16h*)(brow1 + k0 + bkb);
        acc00 = __builtin_amdgcn_wmma_f32_16x16x32_f16(false, a0, false, b0,
                                                       (short)0, acc00, false, false);
        acc01 = __builtin_amdgcn_wmma_f32_16x16x32_f16(false, a0, false, b1,
                                                       (short)0, acc01, false, false);
        acc10 = __builtin_amdgcn_wmma_f32_16x16x32_f16(false, a1, false, b0,
                                                       (short)0, acc10, false, false);
        acc11 = __builtin_amdgcn_wmma_f32_16x16x32_f16(false, a1, false, b1,
                                                       (short)0, acc11, false, false);
    }
    const float bv0 = bias ? bias[n0 + l15]      : 0.0f;
    const float bv1 = bias ? bias[n0 + 16 + l15] : 0.0f;
#pragma unroll
    for (int v = 0; v < 8; ++v) {
        int r0 = m0 + hl * 8 + v;
        int r1 = r0 + 16;
        C[(size_t)r0 * ldc + (n0 + l15)]      = acc00[v] + bv0;
        C[(size_t)r0 * ldc + (n0 + 16 + l15)] = acc01[v] + bv1;
        C[(size_t)r1 * ldc + (n0 + l15)]      = acc10[v] + bv0;
        C[(size_t)r1 * ldc + (n0 + 16 + l15)] = acc11[v] + bv1;
    }
}

// ---------------------------------------------------------------------------
// Sequential LSTM recurrence. One block per direction, 1024 threads.
// gates_x already holds x@Wih^T + bih + bhh.  h,c live in LDS; Whh (f32,
// 2048x512 = 4MB/dir) stays L2-resident across the 512 steps.
// Writes h_t (f16) into out[t][colOff .. colOff+511].
// ---------------------------------------------------------------------------
__global__ __launch_bounds__(1024)
void lstm_seq(const float* __restrict__ gates_x, const float* __restrict__ Whh,
              _Float16* __restrict__ out, int outLd) {
    const int d   = blockIdx.x;          // 0 = forward, 1 = backward
    const int tid = threadIdx.x;
    __shared__ float h_s[HDIM];
    __shared__ float c_s[HDIM];
    __shared__ float g_s[4 * HDIM];
    if (tid < HDIM) { h_s[tid] = 0.0f; c_s[tid] = 0.0f; }
    __syncthreads();

    const float* gx = gates_x + (size_t)d * SLEN * 4 * HDIM;
    const float* W  = Whh     + (size_t)d * 4 * HDIM * HDIM;
    const int colOff = d * HDIM;

    for (int step = 0; step < SLEN; ++step) {
        const int t = d ? (SLEN - 1 - step) : step;
        const float* grow = gx + (size_t)t * 4 * HDIM;
        // matvec: rows tid and tid+1024 of Whh @ h
        float a0 = grow[tid], a1 = grow[tid + 2 * HDIM];
        const float* w0 = W + (size_t)tid * HDIM;
        const float* w1 = W + (size_t)(tid + 2 * HDIM) * HDIM;
        for (int k = 0; k < HDIM; ++k) {
            float hk = h_s[k];
            a0 = fmaf(w0[k], hk, a0);
            a1 = fmaf(w1[k], hk, a1);
        }
        g_s[tid] = a0; g_s[tid + 2 * HDIM] = a1;
        __syncthreads();
        if (tid < HDIM) {
            float ig = g_s[tid], fg = g_s[HDIM + tid];
            float gg = g_s[2 * HDIM + tid], og = g_s[3 * HDIM + tid];
            float c = sigf(fg) * c_s[tid] + sigf(ig) * tanhf(gg);
            float h = sigf(og) * tanhf(c);
            c_s[tid] = c; h_s[tid] = h;
            out[(size_t)t * outLd + colOff + tid] = (_Float16)h;
        }
        __syncthreads();
    }
}

// dtab[34][512] = dist_emb[34,64] @ dist_W^T + dist_b  (tiny; replaces a
// 262144-row GEMV with a 34-row table -> ~7700x less work)
__global__ void dist_kernel(const float* __restrict__ de, const float* __restrict__ dW,
                            const float* __restrict__ db, float* __restrict__ dtab) {
    int b = blockIdx.x, j = threadIdx.x;
    float acc = db[j];
    for (int k = 0; k < 64; ++k)
        acc = fmaf(de[b * 64 + k], dW[j * 64 + k], acc);
    dtab[b * HDIM + j] = acc;
}

// ---------------------------------------------------------------------------
// Scoring: one wave per part. 8 waves/block. Gathered tables are L2-resident
// (proj is 5MB, dtab 68KB; L2 = 192MB). Transcendental-bound; wave32 shfl.
// ---------------------------------------------------------------------------
__global__ __launch_bounds__(256)
void arc_score(const float* __restrict__ proj, const float* __restrict__ dtab,
               const int* __restrict__ ah, const int* __restrict__ am,
               const float* __restrict__ w, float* __restrict__ out, int n) {
    int lane = threadIdx.x & 31;
    int a = blockIdx.x * 8 + (threadIdx.x >> 5);
    if (a >= n) return;
    int h = ah[a], m = am[a];
    int absd = m > h ? m - h : h - m;
    int bin = absd < 10 ? absd : (absd < 40 ? 10 + (absd - 10) / 5 : 16);
    int di = (m > h) ? bin : bin + NBINS;
    const float* hp = proj + (size_t)h * 2560;            // heads: cols 0..511
    const float* mp = proj + (size_t)m * 2560 + 512;      // mods:  cols 512..1023
    const float* dp = dtab + (size_t)di * HDIM;
    float acc = 0.0f;
    for (int k = lane; k < HDIM; k += 32)
        acc = fmaf(tanhf(hp[k] + mp[k] + dp[k]), w[k], acc);
    for (int off = 16; off; off >>= 1) acc += __shfl_xor(acc, off, 32);
    if (lane == 0) out[a] = acc;
}

__global__ __launch_bounds__(256)
void sib_score(const float* __restrict__ proj,
               const int* __restrict__ ih, const int* __restrict__ im,
               const int* __restrict__ is, const float* __restrict__ w,
               float* __restrict__ out, int n) {
    int lane = threadIdx.x & 31;
    int a = blockIdx.x * 8 + (threadIdx.x >> 5);
    if (a >= n) return;
    const float* hp = proj + (size_t)ih[a] * 2560 + 1024; // sib_head
    const float* mp = proj + (size_t)im[a] * 2560 + 1536; // sib_mod
    const float* sp = proj + (size_t)is[a] * 2560 + 2048; // sib_sib
    float acc = 0.0f;
    for (int k = lane; k < HDIM; k += 32)
        acc = fmaf(tanhf(hp[k] + mp[k] + sp[k]), w[k], acc);
    for (int off = 16; off; off >>= 1) acc += __shfl_xor(acc, off, 32);
    if (lane == 0) out[a] = acc;
}

// ---------------------------------------------------------------------------
extern "C" void kernel_launch(void* const* d_in, const int* in_sizes, int n_in,
                              void* d_out, int out_size, void* d_ws, size_t ws_size,
                              hipStream_t stream) {
    const int*   words    = (const int*)  d_in[0];
    const int*   tags     = (const int*)  d_in[1];
    const int*   arc_head = (const int*)  d_in[2];
    const int*   arc_mod  = (const int*)  d_in[3];
    const int*   sib_head = (const int*)  d_in[4];
    const int*   sib_mod  = (const int*)  d_in[5];
    const int*   sib_sib  = (const int*)  d_in[6];
    const float* word_emb = (const float*)d_in[7];
    const float* tag_emb  = (const float*)d_in[8];
    const float* dist_emb = (const float*)d_in[9];
    const float* Wih0     = (const float*)d_in[10];
    const float* Whh0     = (const float*)d_in[11];
    const float* bih0     = (const float*)d_in[12];
    const float* bhh0     = (const float*)d_in[13];
    const float* Wih1     = (const float*)d_in[14];
    const float* Whh1     = (const float*)d_in[15];
    const float* bih1     = (const float*)d_in[16];
    const float* bhh1     = (const float*)d_in[17];
    const float* projWsrc[5] = { (const float*)d_in[18], (const float*)d_in[19],
                                 (const float*)d_in[20], (const float*)d_in[21],
                                 (const float*)d_in[22] };
    const float* dist_W   = (const float*)d_in[23];
    const float* dist_b   = (const float*)d_in[24];
    const float* arc_w    = (const float*)d_in[25];
    const int A    = in_sizes[2];
    const int ASIB = in_sizes[4];

    // workspace layout (bytes), ~33MB total
    char* ws = (char*)d_ws;
    _Float16* x0      = (_Float16*)(ws + 0);            //  512*384  f16
    _Float16* wih0h   = (_Float16*)(ws + 393216);       // 4096*384  f16
    _Float16* wih1h   = (_Float16*)(ws + 3538944);      // 4096*1024 f16
    _Float16* projWh  = (_Float16*)(ws + 11927552);     // 2560*1024 f16
    float*    gatesx  = (float*)   (ws + 17170432);     // 2*512*2048 f32
    float*    cbias   = (float*)   (ws + 25559040);     // 4096 f32
    _Float16* x1      = (_Float16*)(ws + 25575424);     //  512*1024 f16
    _Float16* states  = (_Float16*)(ws + 26624000);     //  512*1024 f16
    float*    proj    = (float*)   (ws + 27672576);     //  512*2560 f32
    float*    dtab    = (float*)   (ws + 32915456);     //   34*512  f32
    float*    out     = (float*)   d_out;

    // 1) embeddings -> x0 f16 [512,384]
    embed_kernel<<<SLEN, 256, 0, stream>>>(words, tags, word_emb, tag_emb, x0);

    // 2) f16 weight copies (padded K for layer 0)
    cvt_kernel<<<4096, 256, 0, stream>>>(Wih0, wih0h, 364, 384);
    cvt_kernel<<<4096, 256, 0, stream>>>(Wih1, wih1h, 1024, 1024);
    for (int i = 0; i < 5; ++i)
        cvt_kernel<<<512, 256, 0, stream>>>(projWsrc[i], projWh + (size_t)i * 512 * 1024,
                                            1024, 1024);

    // 3) layer 0: gates_x = x0 @ Wih0^T + (bih0+bhh0)   (WMMA, tile 64x128)
    bias_kernel<<<16, 256, 0, stream>>>(bih0, bhh0, cbias, 4096);
    for (int d = 0; d < 2; ++d)
        wmma_gemm<<<dim3(2048 / 128, SLEN / 64), 256, 0, stream>>>(
            x0, wih0h + (size_t)d * 2048 * 384, cbias + d * 2048,
            gatesx + (size_t)d * SLEN * 2048, 384, 384, 384, 2048);
    // 4) layer 0 recurrence -> x1 f16 [512,1024]
    lstm_seq<<<2, 1024, 0, stream>>>(gatesx, Whh0, x1, 1024);

    // 5) layer 1: gates_x = x1 @ Wih1^T + (bih1+bhh1)   (WMMA)
    bias_kernel<<<16, 256, 0, stream>>>(bih1, bhh1, cbias, 4096);
    for (int d = 0; d < 2; ++d)
        wmma_gemm<<<dim3(2048 / 128, SLEN / 64), 256, 0, stream>>>(
            x1, wih1h + (size_t)d * 2048 * 1024, cbias + d * 2048,
            gatesx + (size_t)d * SLEN * 2048, 1024, 1024, 1024, 2048);
    // 6) layer 1 recurrence -> states f16 [512,1024]
    lstm_seq<<<2, 1024, 0, stream>>>(gatesx, Whh1, states, 1024);

    // 7) all 5 projections in one WMMA GEMM: [512,1024] x [2560,1024]^T
    wmma_gemm<<<dim3(2560 / 128, SLEN / 64), 256, 0, stream>>>(
        states, projWh, nullptr, proj, 1024, 1024, 1024, 2560);

    // 8) distance table [34,512]
    dist_kernel<<<2 * NBINS, HDIM, 0, stream>>>(dist_emb, dist_W, dist_b, dtab);

    // 9) scores
    arc_score<<<(A + 7) / 8, 256, 0, stream>>>(proj, dtab, arc_head, arc_mod,
                                               arc_w, out, A);
    sib_score<<<(ASIB + 7) / 8, 256, 0, stream>>>(proj, sib_head, sib_mod, sib_sib,
                                                  arc_w, out + A, ASIB);
}
